// ModifiedDWBlock_50242527428694
// MI455X (gfx1250) — matrix operand order repaired and checked by reading, not compiled, not run on hardware
//
#include <hip/hip_runtime.h>
#include <hip/hip_bf16.h>
#include <math.h>

// ---- problem constants (match reference setup_inputs) ----
#define NB   8
#define NT   1024
#define ND   512
#define NFF  2048
#define NH   8
#define DH   64
#define NBT  (NB*NT)

typedef __attribute__((ext_vector_type(16))) _Float16 v16h;
typedef __attribute__((ext_vector_type(8)))  _Float16 v8h;
typedef __attribute__((ext_vector_type(8)))  float    v8f;
typedef __attribute__((address_space(3)))    _Float16 lds_f16_t;

// ---------------- wave32 reductions ----------------
__device__ __forceinline__ float wred_sum(float v){
#pragma unroll
  for (int o = 16; o > 0; o >>= 1) v += __shfl_xor(v, o, 32);
  return v;
}
__device__ __forceinline__ float wred_max(float v){
#pragma unroll
  for (int o = 16; o > 0; o >>= 1) v = fmaxf(v, __shfl_xor(v, o, 32));
  return v;
}
__device__ __forceinline__ float wred_min(float v){
#pragma unroll
  for (int o = 16; o > 0; o >>= 1) v = fminf(v, __shfl_xor(v, o, 32));
  return v;
}

// ---------------- elementwise helpers ----------------
__global__ void k_cvt(const float* __restrict__ in, _Float16* __restrict__ out, int n){
  int i = blockIdx.x * blockDim.x + threadIdx.x;
  if (i < n) out[i] = (_Float16)in[i];
}

// in: (K x N) f32 row-major  ->  out: (N x K) f16 row-major (B^T for WMMA B loads)
__global__ void k_cvt_tr(const float* __restrict__ in, _Float16* __restrict__ out, int K, int N){
  int i = blockIdx.x * blockDim.x + threadIdx.x;
  if (i < K * N){
    int n = i / K, k = i % K;
    out[i] = (_Float16)in[(size_t)k * N + n];
  }
}

__global__ void k_zero(float* __restrict__ p, int n){
  int i = blockIdx.x * blockDim.x + threadIdx.x;
  if (i < n) p[i] = 0.f;
}

// vth[b][d][t] = qkvh[2][b][t][d]   (transpose V so PV B-operand loads are contiguous)
__global__ void k_vtr(const _Float16* __restrict__ qkvh, _Float16* __restrict__ vth, int n){
  int i = blockIdx.x * blockDim.x + threadIdx.x;
  if (i < n){
    int b = i / (ND * NT);
    int r = i % (ND * NT);
    int d = r / NT;
    int t = r % NT;
    vth[i] = qkvh[(size_t)2 * NBT * ND + (size_t)b * NT * ND + (size_t)t * ND + d];
  }
}

// ---------------- WMMA GEMM: C(MxN) = A(MxK) * B(KxN), Bt is B^T (NxK) f16 ----------------
// 4 waves per block; 64x64 output tile; B panel double-buffered in LDS via
// global_load_async_to_lds_b128 and shared by all 4 waves. f32 accumulate.
__global__ void __launch_bounds__(128) k_gemm(
                       const _Float16* __restrict__ A, const _Float16* __restrict__ Bt,
                       float* __restrict__ Cf, _Float16* __restrict__ Ch,
                       int N, int K, int relu, int outHalf){
  __shared__ __align__(32) _Float16 pan[2][64][32];   // 2 x (64 cols x 32 k) = 8 KB
  const int tid  = threadIdx.x;
  const int wv   = tid >> 5;
  const int lane = tid & 31;
  const int hf   = lane >> 4;
  const int l15  = lane & 15;
  const int m0 = blockIdx.y * 64 + wv * 16;   // this wave's 16-row strip
  const int n0 = blockIdx.x * 64;

  // this thread's two b128 chunks of the 64x32 panel (manually unrolled)
  const int c0 = tid,        r0 = c0 >> 2, q0 = c0 & 3;
  const int c1 = tid + 128,  r1 = c1 >> 2, q1 = c1 & 3;
  const _Float16* src0 = Bt + (size_t)(n0 + r0) * K + q0 * 8;
  const _Float16* src1 = Bt + (size_t)(n0 + r1) * K + q1 * 8;
  unsigned dst0[2], dst1[2];
#pragma unroll
  for (int bf = 0; bf < 2; bf++){
    dst0[bf] = (unsigned)(size_t)(lds_f16_t*)&pan[bf][r0][q0 * 8];
    dst1[bf] = (unsigned)(size_t)(lds_f16_t*)&pan[bf][r1][q1 * 8];
  }
#define ISSUE_PANEL(k0, bf)                                                        \
  do {                                                                             \
    asm volatile("global_load_async_to_lds_b128 %0, %1, off"                       \
                 :: "v"(dst0[bf]),                                                 \
                    "v"((unsigned long long)(src0 + (k0))) : "memory");            \
    asm volatile("global_load_async_to_lds_b128 %0, %1, off"                       \
                 :: "v"(dst1[bf]),                                                 \
                    "v"((unsigned long long)(src1 + (k0))) : "memory");            \
  } while (0)

  v8f acc[4] = {};
  const _Float16* arow = A + (size_t)(m0 + l15) * K;
  const int nk = K / 32;

  ISSUE_PANEL(0, 0);
  for (int i = 0; i < nk; i++){
    const int bf = i & 1;
    if (i + 1 < nk){
      ISSUE_PANEL((i + 1) * 32, (i + 1) & 1);
      // async loads complete in order: <=2 outstanding means panel i has landed,
      // while panel i+1 keeps streaming under the WMMA work below.
      asm volatile("s_wait_asynccnt 2" ::: "memory");
    } else {
      asm volatile("s_wait_asynccnt 0" ::: "memory");
    }
    __syncthreads();

    const int k0 = i * 32;
    // A 16x32 f16 fragment: lane group hf holds K chunk [hf*16,+16)
    v8h alo = *(const v8h*)(arow + k0 + hf * 16);
    v8h ahi = *(const v8h*)(arow + k0 + hf * 16 + 8);
    v16h a;
#pragma unroll
    for (int e = 0; e < 8; e++){ a[e] = alo[e]; a[8 + e] = ahi[e]; }
#pragma unroll
    for (int nt = 0; nt < 4; nt++){
      // B 32x16 fragment: lane = column n, elems = 16 consecutive k, from LDS panel
      v16h b = *(const v16h*)&pan[bf][nt * 16 + l15][hf * 16];
      acc[nt] = __builtin_amdgcn_wmma_f32_16x16x32_f16(false, a, false, b,
                                                       (short)0, acc[nt], false, false);
    }
    __syncthreads();   // all waves done with panel i before buffer bf is overwritten
  }
#undef ISSUE_PANEL

#pragma unroll
  for (int nt = 0; nt < 4; nt++){
#pragma unroll
    for (int r = 0; r < 8; r++){
      float v = acc[nt][r];
      if (relu) v = fmaxf(v, 0.f);
      size_t idx = (size_t)(m0 + r + 8 * hf) * N + n0 + nt * 16 + l15;
      if (outHalf) Ch[idx] = (_Float16)v; else Cf[idx] = v;
    }
  }
}

// ---------------- fused attention: one wave per (b, h, 16-query tile) ----------------
// scores (QK^T/8, optional block-diag mask) -> LDS, row softmax, column sums -> wacc,
// ctx = P @ V via WMMA.
__global__ void k_attn(const _Float16* __restrict__ qkvh, const _Float16* __restrict__ vth,
                       const int* __restrict__ ids, float* __restrict__ wacc,
                       float* __restrict__ ctx, int useMask){
  __shared__ float sc[16][NT];       // 64 KB of the 320 KB/WGP LDS
  const int lane = threadIdx.x;
  const int hf   = lane >> 4;
  const int l15  = lane & 15;
  const int qt = blockIdx.x;
  const int h  = blockIdx.y;
  const int b  = blockIdx.z;

  const _Float16* Qb = qkvh + (size_t)b * NT * ND + h * DH;
  const _Float16* Kb = qkvh + (size_t)NBT * ND + (size_t)b * NT * ND + h * DH;
  const _Float16* Vt = vth + (size_t)b * ND * NT + (size_t)(h * DH) * NT;

  // preload both K-chunks of the Q fragment (reused across all key tiles)
  const _Float16* qrow = Qb + (size_t)(qt * 16 + l15) * ND;
  v16h aq[2];
#pragma unroll
  for (int c = 0; c < 2; c++){
    v8h lo = *(const v8h*)(qrow + c * 32 + hf * 16);
    v8h hi = *(const v8h*)(qrow + c * 32 + hf * 16 + 8);
#pragma unroll
    for (int e = 0; e < 8; e++){ aq[c][e] = lo[e]; aq[c][8 + e] = hi[e]; }
  }

  // ---- scores ----
  for (int kt = 0; kt < NT / 16; kt++){
    v8f acc = {};
#pragma unroll
    for (int c = 0; c < 2; c++){
      v16h bk = *(const v16h*)(Kb + (size_t)(kt * 16 + l15) * ND + c * 32 + hf * 16);
      acc = __builtin_amdgcn_wmma_f32_16x16x32_f16(false, aq[c], false, bk,
                                                   (short)0, acc, false, false);
    }
#pragma unroll
    for (int r = 0; r < 8; r++){
      int m = r + 8 * hf;
      float s = acc[r] * 0.125f;   // 1/sqrt(dh)
      if (useMask && (ids[b * NT + qt * 16 + m] != ids[b * NT + kt * 16 + l15]))
        s = -__builtin_inff();
      sc[m][kt * 16 + l15] = s;
    }
  }
  __syncthreads();

  // ---- row softmax (diagonal is always unmasked, so sum > 0) ----
  for (int r = 0; r < 16; r++){
    float mx = -__builtin_inff();
    for (int c = lane; c < NT; c += 32) mx = fmaxf(mx, sc[r][c]);
    mx = wred_max(mx);
    float sum = 0.f;
    for (int c = lane; c < NT; c += 32){
      float e = __expf(sc[r][c] - mx);
      sc[r][c] = e; sum += e;
    }
    sum = wred_sum(sum);
    float inv = 1.f / sum;
    for (int c = lane; c < NT; c += 32) sc[r][c] *= inv;
  }
  __syncthreads();

  // ---- attention received per key: wacc[b,k] += mean_h sum_q probs ----
  for (int c = lane; c < NT; c += 32){
    float s = 0.f;
#pragma unroll
    for (int r = 0; r < 16; r++) s += sc[r][c];
    atomicAdd(&wacc[b * NT + c], s * (1.0f / NH));
  }

  // ---- ctx = P @ V ----
  v8f o[4] = {};
  for (int kc = 0; kc < NT / 32; kc++){
    v16h a;
#pragma unroll
    for (int e = 0; e < 8; e++) a[e]     = (_Float16)sc[l15][kc * 32 + hf * 16 + e];
#pragma unroll
    for (int e = 0; e < 8; e++) a[8 + e] = (_Float16)sc[l15][kc * 32 + hf * 16 + 8 + e];
#pragma unroll
    for (int nt = 0; nt < 4; nt++){
      v16h bv = *(const v16h*)(Vt + (size_t)(nt * 16 + l15) * NT + kc * 32 + hf * 16);
      o[nt] = __builtin_amdgcn_wmma_f32_16x16x32_f16(false, a, false, bv,
                                                     (short)0, o[nt], false, false);
    }
  }
  float* crow = ctx + (size_t)b * NT * ND;
#pragma unroll
  for (int nt = 0; nt < 4; nt++)
#pragma unroll
    for (int r = 0; r < 8; r++)
      crow[(size_t)(qt * 16 + r + 8 * hf) * ND + h * DH + nt * 16 + l15] = o[nt][r];
}

// ---------------- residual + LayerNorm: y = LN(x + t), one wave per row of 512 ----------------
__global__ void k_addln(const float* __restrict__ x, const float* __restrict__ t,
                        float* __restrict__ y){
  const int lane = threadIdx.x;
  const size_t row = blockIdx.x;
  float loc[ND / 32];
  float s = 0.f;
#pragma unroll
  for (int i = 0; i < ND / 32; i++){
    float v = x[row * ND + lane + i * 32] + t[row * ND + lane + i * 32];
    loc[i] = v; s += v;
  }
  s = wred_sum(s);
  float mean = s * (1.0f / ND);
  float vs = 0.f;
#pragma unroll
  for (int i = 0; i < ND / 32; i++){ float d = loc[i] - mean; vs += d * d; }
  vs = wred_sum(vs);
  float rstd = rsqrtf(vs * (1.0f / ND) + 1e-5f);
#pragma unroll
  for (int i = 0; i < ND / 32; i++)
    y[row * ND + lane + i * 32] = (loc[i] - mean) * rstd;
}

// ---------------- min-max normalize + threshold (per batch, one wave) ----------------
__global__ void k_mmth(const float* __restrict__ wacc, int* __restrict__ wbool){
  int b = blockIdx.x, lane = threadIdx.x;
  float mn = __builtin_inff(), mx = -__builtin_inff();
  for (int c = lane; c < NT; c += 32){
    float v = wacc[b * NT + c];
    mn = fminf(mn, v); mx = fmaxf(mx, v);
  }
  mn = wred_min(mn); mx = wred_max(mx);
  float inv = 1.f / (mx - mn + 1e-8f);
  for (int c = lane; c < NT; c += 32){
    float v = (wacc[b * NT + c] - mn) * inv;
    wbool[b * NT + c] = (v >= 0.5f) ? 1 : 0;
  }
}

// ---------------- sequential window-id scan (length-1 runs absorbed) ----------------
__global__ void k_wids(const int* __restrict__ wbool, int* __restrict__ ids){
  int b = threadIdx.x;
  if (b < NB){
    const int* w = wbool + b * NT;
    int* id = ids + b * NT;
    int cur = w[0], start = 0, wid = 0;
    id[0] = 0;
    for (int t = 1; t < NT; t++){
      int wt = w[t];
      if (wt != cur){
        if (start + 1 == t){ cur = wt; }           // merge: length-1 run absorbed
        else { cur = wt; start = t; wid++; }       // close window, open new
      }
      id[t] = wid;
    }
  }
}

// ---------------- window_mapping[b,k,j] = (ids[b,j]==k) ----------------
__global__ void k_map(const int* __restrict__ ids, float* __restrict__ map){
  int k = blockIdx.x, b = blockIdx.y;
  for (int j = threadIdx.x; j < NT; j += blockDim.x)
    map[(size_t)b * NT * NT + (size_t)k * NT + j] = (ids[b * NT + j] == k) ? 1.f : 0.f;
}

// ---------------- softmax over T per batch (wl), one wave ----------------
__global__ void k_softb(const float* __restrict__ in, float* __restrict__ out){
  int b = blockIdx.x, lane = threadIdx.x;
  float mx = -__builtin_inff();
  for (int c = lane; c < NT; c += 32) mx = fmaxf(mx, in[b * NT + c]);
  mx = wred_max(mx);
  float s = 0.f;
  for (int c = lane; c < NT; c += 32) s += __expf(in[b * NT + c] - mx);
  s = wred_sum(s);
  float inv = 1.f / s;
  for (int c = lane; c < NT; c += 32) out[b * NT + c] = __expf(in[b * NT + c] - mx) * inv;
}

// ---------------- d_out region helpers ----------------
__global__ void k_zero_word(float* __restrict__ out){
  size_t i = (size_t)blockIdx.x * blockDim.x + threadIdx.x;
  if (i < (size_t)NB * NT * ND){
    size_t b = i / ((size_t)NT * ND), r = i % ((size_t)NT * ND);
    out[b * 2 * NT * ND + r] = 0.f;
  }
}
__global__ void k_pool(const float* __restrict__ out_l, const float* __restrict__ wl,
                       const int* __restrict__ ids, float* __restrict__ out){
  int t = blockIdx.x % NT, b = blockIdx.x / NT;
  int w = ids[b * NT + t];
  float scale = wl[b * NT + t];
  const float* src = out_l + ((size_t)b * NT + t) * ND;
  float* dst = out + (size_t)b * 2 * NT * ND + (size_t)w * ND;
  for (int d = threadIdx.x; d < ND; d += blockDim.x)
    atomicAdd(&dst[d], src[d] * scale);
}
__global__ void k_copyx(const float* __restrict__ x, float* __restrict__ out){
  size_t i = (size_t)blockIdx.x * blockDim.x + threadIdx.x;
  if (i < (size_t)NB * NT * ND){
    size_t b = i / ((size_t)NT * ND), r = i % ((size_t)NT * ND);
    out[b * 2 * NT * ND + (size_t)NT * ND + r] = x[i];
  }
}

// =====================================================================
extern "C" void kernel_launch(void* const* d_in, const int* in_sizes, int n_in,
                              void* d_out, int out_size, void* d_ws, size_t ws_size,
                              hipStream_t stream){
  const float* x      = (const float*)d_in[0];
  const float* wsrc[8] = { (const float*)d_in[1], (const float*)d_in[2],
                           (const float*)d_in[3], (const float*)d_in[4],
                           (const float*)d_in[5], (const float*)d_in[6],
                           (const float*)d_in[7], (const float*)d_in[8] };

  // ---- workspace carve-out ----
  size_t off = 0;
  auto alloc = [&](size_t bytes)->char*{
    char* p = (char*)d_ws + off;
    off += (bytes + 255) & ~(size_t)255;
    return p;
  };
  _Float16* xh    = (_Float16*)alloc((size_t)NBT * ND * 2);
  const size_t WPL = (size_t)3 * ND * ND + (size_t)ND * ND + (size_t)ND * NFF + (size_t)NFF * ND;
  _Float16* wh    = (_Float16*)alloc(2 * WPL * 2);           // transposed f16 weights, both layers
  _Float16* qkvh  = (_Float16*)alloc((size_t)3 * NBT * ND * 2);
  _Float16* vth   = (_Float16*)alloc((size_t)NB * ND * NT * 2);
  float*    ctx   = (float*)   alloc((size_t)NBT * ND * 4);
  _Float16* ctxh  = (_Float16*)alloc((size_t)NBT * ND * 2);
  float*    tmpD  = (float*)   alloc((size_t)NBT * ND * 4);
  float*    x1    = (float*)   alloc((size_t)NBT * ND * 4);
  _Float16* x1h   = (_Float16*)alloc((size_t)NBT * ND * 2);
  _Float16* ffh   = (_Float16*)alloc((size_t)NBT * NFF * 2);
  float*    out_v = (float*)   alloc((size_t)NBT * ND * 4);
  float*    out_l = (float*)   alloc((size_t)NBT * ND * 4);
  float*    w_acc = (float*)   alloc((size_t)NBT * 4);
  float*    wlacc = (float*)   alloc((size_t)NBT * 4);
  float*    wl    = (float*)   alloc((size_t)NBT * 4);
  int*      wboolb= (int*)     alloc((size_t)NBT * 4);
  int*      ids   = (int*)     alloc((size_t)NBT * 4);
  (void)ws_size; (void)in_sizes; (void)n_in; (void)out_size;

  float* outc = (float*)d_out;                         // (B, 2T, D)
  float* omap = outc + (size_t)NB * 2 * NT * ND;       // (B, T, T)

  const int EB = 256;
  auto blocks = [&](size_t n){ return (unsigned)((n + EB - 1) / EB); };

  // ---- convert + transpose all weights (f32 -> f16, B^T layout) ----
  for (int L = 0; L < 2; L++){
    _Float16* wb = wh + (size_t)L * WPL;
    const float* qkvw = wsrc[L * 4 + 0];
    for (int c = 0; c < 3; c++)
      k_cvt_tr<<<blocks((size_t)ND*ND), EB, 0, stream>>>(qkvw + (size_t)c*ND*ND,
                                                         wb + (size_t)c*ND*ND, ND, ND);
    k_cvt_tr<<<blocks((size_t)ND*ND), EB, 0, stream>>>(wsrc[L*4+1], wb + (size_t)3*ND*ND, ND, ND);
    k_cvt_tr<<<blocks((size_t)ND*NFF), EB, 0, stream>>>(wsrc[L*4+2], wb + (size_t)4*ND*ND, ND, NFF);
    k_cvt_tr<<<blocks((size_t)NFF*ND), EB, 0, stream>>>(wsrc[L*4+3],
                                                        wb + (size_t)4*ND*ND + (size_t)NFF*ND, NFF, ND);
  }

  const float* layer_in_f32[2];
  float* layer_out_f32[2] = { out_v, out_l };
  float* layer_wacc[2]    = { w_acc, wlacc };
  layer_in_f32[0] = x;
  layer_in_f32[1] = out_v;

  for (int L = 0; L < 2; L++){
    _Float16* wb   = wh + (size_t)L * WPL;
    _Float16* qkvT = wb;
    _Float16* outT = wb + (size_t)3 * ND * ND;
    _Float16* w1T  = wb + (size_t)4 * ND * ND;
    _Float16* w2T  = w1T + (size_t)NFF * ND;

    // x (f32) -> f16
    k_cvt<<<blocks((size_t)NBT*ND), EB, 0, stream>>>(layer_in_f32[L], xh, NBT * ND);

    // QKV GEMMs: (8192x512) @ (512x512) -> f16
    for (int c = 0; c < 3; c++)
      k_gemm<<<dim3(ND/64, NBT/64), 128, 0, stream>>>(xh, qkvT + (size_t)c*ND*ND,
                                                      nullptr, qkvh + (size_t)c*NBT*ND,
                                                      ND, ND, 0, 1);
    // transpose V per (b): vth[b][d][t]
    k_vtr<<<blocks((size_t)NB*ND*NT), EB, 0, stream>>>(qkvh, vth, NB * ND * NT);

    // attention (layer 0: no mask, accumulate w; layer 1: block-diag mask, accumulate wl)
    k_zero<<<blocks((size_t)NBT), EB, 0, stream>>>(layer_wacc[L], NBT);
    k_attn<<<dim3(NT/16, NH, NB), 32, 0, stream>>>(qkvh, vth, ids, layer_wacc[L], ctx, L);

    // out projection + residual + LN
    k_cvt<<<blocks((size_t)NBT*ND), EB, 0, stream>>>(ctx, ctxh, NBT * ND);
    k_gemm<<<dim3(ND/64, NBT/64), 128, 0, stream>>>(ctxh, outT, tmpD, nullptr, ND, ND, 0, 0);
    k_addln<<<NBT, 32, 0, stream>>>(layer_in_f32[L], tmpD, x1);

    // FFN: relu(x1 @ w1) @ w2, residual + LN
    k_cvt<<<blocks((size_t)NBT*ND), EB, 0, stream>>>(x1, x1h, NBT * ND);
    k_gemm<<<dim3(NFF/64, NBT/64), 128, 0, stream>>>(x1h, w1T, nullptr, ffh, NFF, ND, 1, 1);
    k_gemm<<<dim3(ND/64, NBT/64), 128, 0, stream>>>(ffh, w2T, tmpD, nullptr, ND, NFF, 0, 0);
    k_addln<<<NBT, 32, 0, stream>>>(x1, tmpD, layer_out_f32[L]);

    if (L == 0){
      // window mask generation between the two layers
      k_mmth<<<NB, 32, 0, stream>>>(w_acc, wboolb);
      k_wids<<<1, 32, 0, stream>>>(wboolb, ids);
      k_map<<<dim3(NT, NB), 256, 0, stream>>>(ids, omap);
    }
  }

  // wl = softmax over tokens of summed layer-2 attention
  k_softb<<<NB, 32, 0, stream>>>(wlacc, wl);

  // window-weighted pooling + copy of original x into the concat output
  k_zero_word<<<blocks((size_t)NB*NT*ND), EB, 0, stream>>>(outc);
  k_pool<<<NB * NT, 128, 0, stream>>>(out_l, wl, ids, outc);
  k_copyx<<<blocks((size_t)NB*NT*ND), EB, 0, stream>>>(x, outc);
}
